// ScaledDotProductAttention_6923487281447
// MI455X (gfx1250) — matrix operand order repaired and checked by reading, not compiled, not run on hardware
//
#include <hip/hip_runtime.h>
#include <cstdint>
#include <cstddef>

typedef __attribute__((ext_vector_type(16))) _Float16 v16h;
typedef __attribute__((ext_vector_type(8)))  float    v8f;

#define S_LEN   2048
#define DKDIM   64
#define TQ      16
#define QTILES  (S_LEN / TQ)     /* 128 */
#define KCHUNKS (S_LEN / 32)     /* 64  */
#define NWAVES  8
#define SCALE   0.044194173824159216f  /* 1/sqrt(512) ; TEMPER uses D_MODEL=512 */

__global__ __launch_bounds__(256)
void attn_fwd_wmma_kernel(const float* __restrict__ q,
                          const float* __restrict__ k,
                          const float* __restrict__ v,
                          const int*   __restrict__ mask,
                          float* __restrict__ out,    // [B, S, DK]
                          float* __restrict__ attn)   // [B, S, S]
{
    extern __shared__ float smem[];
    float* sc   = smem;                      // 16 x 2048 score strip (128 KiB)
    float* pv   = smem + TQ * S_LEN;         // 16 x 64 PV reduction
    float* redA = pv + TQ * DKDIM;           // 256 partial max
    float* redB = redA + 256;                // 256 partial sum

    const int tid  = threadIdx.x;
    const int lane = tid & 31;
    const int wid  = tid >> 5;
    const int bIdx = blockIdx.x / QTILES;
    const int qt   = blockIdx.x % QTILES;
    const int q0   = qt * TQ;

    const int n   = lane & 15;   // tile column (B/C/D) or row m (A)
    const int sel = lane >> 4;   // half-wave select

    // ---------------- Load Q fragments (A: 16x32 f16, two K-chunks) ----------
    // A layout: lanes 0-15 row m, VGPR v<4 -> K=2v,2v+1 (+8 for upper half-wave),
    //           VGPR v>=4 -> K=16+2(v-4) (+8 for upper half-wave).
    const size_t qRow = ((size_t)bIdx * S_LEN + q0 + n) * DKDIM;
    v16h aQ0, aQ1;
    #pragma unroll
    for (int vv = 0; vv < 8; ++vv) {
        const int kk = ((vv < 4) ? (2 * vv) : (16 + 2 * (vv - 4))) + sel * 8;
        const float2 f0 = *(const float2*)(q + qRow + kk);
        const float2 f1 = *(const float2*)(q + qRow + kk + 32);
        aQ0[2 * vv]     = (_Float16)f0.x;
        aQ0[2 * vv + 1] = (_Float16)f0.y;
        aQ1[2 * vv]     = (_Float16)f1.x;
        aQ1[2 * vv + 1] = (_Float16)f1.y;
    }

    // ---------------- Phase 1: S = mask(QK^T * scale) -> LDS ------------------
    for (int kt = wid; kt < QTILES; kt += NWAVES) {
        const int kg = kt * TQ;
        // B layout (32x16): lane col n, same K order as A (v<4: 2v,2v+1; v>=4: 16+2(v-4); +8 upper half-wave).
        // B[kk][n] = K[kg+n][kk] -> contiguous float2 along kk.
        const size_t kRow = ((size_t)bIdx * S_LEN + kg + n) * DKDIM;
        v16h bK0, bK1;
        #pragma unroll
        for (int vv = 0; vv < 8; ++vv) {
            const int kk = ((vv < 4) ? (2 * vv) : (16 + 2 * (vv - 4))) + sel * 8;
            const float2 f0 = *(const float2*)(k + kRow + kk);
            const float2 f1 = *(const float2*)(k + kRow + kk + 32);
            bK0[2 * vv]     = (_Float16)f0.x;
            bK0[2 * vv + 1] = (_Float16)f0.y;
            bK1[2 * vv]     = (_Float16)f1.x;
            bK1[2 * vv + 1] = (_Float16)f1.y;
        }
        if (kt + NWAVES < QTILES) {
            __builtin_prefetch(k + kRow + (size_t)NWAVES * TQ * DKDIM, 0, 1);
        }

        v8f acc = {};
        acc = __builtin_amdgcn_wmma_f32_16x16x32_f16(false, aQ0, false, bK0,
                                                     (short)0, acc, false, false);
        acc = __builtin_amdgcn_wmma_f32_16x16x32_f16(false, aQ1, false, bK1,
                                                     (short)0, acc, false, false);

        // C/D layout: VGPR vv -> row m = vv + 8*sel, col = n.
        #pragma unroll
        for (int vv = 0; vv < 8; ++vv) {
            const int m  = vv + sel * 8;
            const int ng = kg + n;
            const int mk = mask[((size_t)bIdx * S_LEN + (q0 + m)) * S_LEN + ng];
            sc[m * S_LEN + ng] = mk ? -__builtin_inff() : acc[vv] * SCALE;
        }
    }
    __syncthreads();

    // ---------------- Phase 2: row softmax over LDS strip ---------------------
    const int row  = tid >> 4;    // 0..15
    const int cofs = tid & 15;
    float* scRow = sc + row * S_LEN;

    float lmax = -__builtin_inff();
    for (int c = cofs; c < S_LEN; c += 16) lmax = fmaxf(lmax, scRow[c]);
    redA[tid] = lmax;
    #pragma unroll
    for (int i = 0; i < 4; ++i) pv[tid + 256 * i] = 0.0f;  // zero PV reduction
    __syncthreads();

    float rmax = -__builtin_inff();
    #pragma unroll
    for (int j = 0; j < 16; ++j) rmax = fmaxf(rmax, redA[(row << 4) + j]);

    float lsum = 0.0f;
    for (int c = cofs; c < S_LEN; c += 16) {
        const float e = __expf(scRow[c] - rmax);
        scRow[c] = e;
        lsum += e;
    }
    redB[tid] = lsum;
    __syncthreads();

    float rsum = 0.0f;
    #pragma unroll
    for (int j = 0; j < 16; ++j) rsum += redB[(row << 4) + j];
    const float rinv = 1.0f / rsum;

    float* attnRow = attn + ((size_t)bIdx * S_LEN + (q0 + row)) * S_LEN;
    for (int c = cofs; c < S_LEN; c += 16) {
        const float p = scRow[c] * rinv;
        scRow[c]   = p;           // keep normalized P in LDS for PV
        attnRow[c] = p;           // coalesced attn output
    }
    __syncthreads();

    // ---------------- Phase 3: O = P x V via WMMA ------------------------------
    v8f accv[4] = {};
    for (int kc = wid; kc < KCHUNKS; kc += NWAVES) {
        const int kk0 = kc * 32;
        // A = P tile (16x32 f16) from LDS, row m = n (lane&15)
        v16h aP;
        const float* pRow = sc + n * S_LEN + kk0;
        #pragma unroll
        for (int vv = 0; vv < 8; ++vv) {
            const int kk = ((vv < 4) ? (2 * vv) : (16 + 2 * (vv - 4))) + sel * 8;
            const float2 f = *(const float2*)(pRow + kk);
            aP[2 * vv]     = (_Float16)f.x;
            aP[2 * vv + 1] = (_Float16)f.y;
        }
        // B = V tile (32 x 16) per output-dim tile nt: B[kk][n] = V[kk0+kk][16*nt+n]
        const size_t vBase = ((size_t)bIdx * S_LEN + kk0) * DKDIM;
        #pragma unroll
        for (int nt = 0; nt < 4; ++nt) {
            v16h bV;
            #pragma unroll
            for (int vv = 0; vv < 8; ++vv) {
                const int kk = ((vv < 4) ? (2 * vv) : (16 + 2 * (vv - 4))) + sel * 8;
                const size_t a0 = vBase + (size_t)kk * DKDIM + nt * 16 + n;
                bV[2 * vv]     = (_Float16)v[a0];
                bV[2 * vv + 1] = (_Float16)v[a0 + DKDIM];
            }
            accv[nt] = __builtin_amdgcn_wmma_f32_16x16x32_f16(false, aP, false, bV,
                                                              (short)0, accv[nt],
                                                              false, false);
        }
    }

    // Cross-wave reduction of partial 16x64 tiles via LDS float atomics
    #pragma unroll
    for (int nt = 0; nt < 4; ++nt) {
        #pragma unroll
        for (int vv = 0; vv < 8; ++vv) {
            const int m = vv + sel * 8;
            atomicAdd(&pv[m * DKDIM + nt * 16 + n], accv[nt][vv]);
        }
    }
    __syncthreads();

    // ---------------- Write output tile ---------------------------------------
    const size_t oBase = ((size_t)bIdx * S_LEN + q0) * DKDIM;
    #pragma unroll
    for (int i = 0; i < 4; ++i) {
        const int idx = tid + 256 * i;      // idx = m*64 + d
        out[oBase + idx] = pv[idx];
    }
}

extern "C" void kernel_launch(void* const* d_in, const int* in_sizes, int n_in,
                              void* d_out, int out_size, void* d_ws, size_t ws_size,
                              hipStream_t stream) {
    (void)n_in; (void)out_size; (void)d_ws; (void)ws_size;
    const float* q    = (const float*)d_in[0];
    const float* k    = (const float*)d_in[1];
    const float* v    = (const float*)d_in[2];
    const int*   mask = (const int*)d_in[3];

    const int B = in_sizes[0] / (S_LEN * DKDIM);   // 32
    float* out  = (float*)d_out;                   // [B,S,DK] first
    float* attn = out + (size_t)B * S_LEN * DKDIM; // then [B,S,S]

    const size_t ldsBytes = (size_t)(TQ * S_LEN + TQ * DKDIM + 512) * sizeof(float);
    attn_fwd_wmma_kernel<<<dim3(B * QTILES), dim3(256), ldsBytes, stream>>>(
        q, k, v, mask, out, attn);
}
